// EdgeConditionedConv_43774306680927
// MI455X (gfx1250) — compile-verified, matrix operand order checked by reading, not run on hardware
//
#include <hip/hip_runtime.h>
#include <hip/hip_bf16.h>
#include <math.h>

#define N_NODES   25000
#define N_EDGES   50000
#define NODE_DIM  64
#define EDGE_DIM  16
#define HID       128
#define E_PAD     50176   /* 196 blocks * 256 edges */
#define BROW      136     /* 128 + 8 pad: row stride 272B = 68 dwords -> conflict-free halves */

typedef __attribute__((ext_vector_type(16))) __bf16 v16bf;
typedef __attribute__((ext_vector_type(8)))  __bf16 v8bf;
typedef __attribute__((ext_vector_type(8)))  float  v8f;

// exact pointer types of __builtin_amdgcn_global_load_async_to_lds_b128
typedef int v4i_vs __attribute__((vector_size(16)));
typedef v4i_vs __attribute__((address_space(1)))* gv4i_p;   // global
typedef v4i_vs __attribute__((address_space(3)))* lv4i_p;   // LDS

#if __has_builtin(__builtin_amdgcn_global_load_async_to_lds_b128)
#define HAVE_ASYNC_LDS 1
#else
#define HAVE_ASYNC_LDS 0
#endif

// ---------------- helpers ----------------
__device__ __forceinline__ v16bf combine16(const __bf16* lo, const __bf16* hi) {
    v8bf a = *(const v8bf*)lo;
    v8bf b = *(const v8bf*)hi;
    v16bf r;
#pragma unroll
    for (int i = 0; i < 8; ++i) { r[i] = a[i]; r[8 + i] = b[i]; }
    return r;
}

__device__ __forceinline__ v16bf scale_bf(v16bf a, float s) {
    v16bf r;
#pragma unroll
    for (int i = 0; i < 16; ++i) r[i] = (__bf16)((float)a[i] * s);
    return r;
}

__device__ __forceinline__ void wait_async0() {
#if HAVE_ASYNC_LDS
#if __has_builtin(__builtin_amdgcn_s_wait_asynccnt)
    __builtin_amdgcn_s_wait_asynccnt(0);
#else
    asm volatile("s_wait_asynccnt 0x0" ::: "memory");
#endif
#endif
}

// ---------------- k0: zero the aggregation buffer (d_out) ----------------
__global__ void zero_kernel(float* __restrict__ p, int n) {
    int i = blockIdx.x * blockDim.x + threadIdx.x;
    if (i < n) p[i] = 0.0f;
}

// ---------------- k1: repack W2 and b2 to bf16 (identity layout!) --------
__global__ void repack_kernel(const float* __restrict__ W2, const float* __restrict__ b2,
                              __bf16* __restrict__ Bw, __bf16* __restrict__ Bb2) {
    int i = blockIdx.x * blockDim.x + threadIdx.x;
    if (i < NODE_DIM * NODE_DIM * HID) Bw[i] = (__bf16)W2[i];
    if (i < NODE_DIM * NODE_DIM)       Bb2[i] = (__bf16)b2[i];
}

// ---------------- k2a: h = gelu(edge_attr @ W1^T + b1), bf16, padded -----
__global__ void hcalc_kernel(const float* __restrict__ ea, const float* __restrict__ W1,
                             const float* __restrict__ b1, __bf16* __restrict__ hbf) {
    int idx = blockIdx.x * blockDim.x + threadIdx.x;     // E_PAD * 128 threads
    int e = idx >> 7, k = idx & 127;
    float v = 0.0f;
    if (e < N_EDGES) {
        v = b1[k];
        const float* er = ea + (size_t)e * EDGE_DIM;
        const float* wr = W1 + (size_t)k * EDGE_DIM;
#pragma unroll
        for (int t = 0; t < EDGE_DIM; ++t) v += er[t] * wr[t];
        v = 0.5f * v * (1.0f + erff(v * 0.70710678118654752f));   // exact gelu
    }
    hbf[idx] = (__bf16)v;
}

// ---------------- k2b: gather x[src] to bf16, padded ---------------------
__global__ void xgather_kernel(const float* __restrict__ x, const int* __restrict__ src,
                               __bf16* __restrict__ xsbf) {
    int idx = blockIdx.x * blockDim.x + threadIdx.x;     // E_PAD * 64 threads
    int e = idx >> 6, j = idx & 63;
    float v = 0.0f;
    if (e < N_EDGES) v = x[(size_t)src[e] * NODE_DIM + j];
    xsbf[idx] = (__bf16)v;
}

// ---------------- k3: fused edge-GEMM (WMMA bf16) + scatter-add ----------
// messages[e,i] = sum_{j,k} W2[i*64+j,k]*xs[e,j]*h[e,k] + sum_j b2[i*64+j]*xs[e,j]
// One wave = 32 edges (2 M-tiles). B fragments are edge-independent -> staged
// once per block per j into double-buffered LDS (8x cut in L2 traffic), using
// async global->LDS copies (ASYNCcnt) when the toolchain exposes them.
__global__ __launch_bounds__(256) void msg_kernel(
    const __bf16* __restrict__ hbf, const __bf16* __restrict__ xsbf,
    const __bf16* __restrict__ Bw,  const __bf16* __restrict__ Bb2,
    const int* __restrict__ dstIdx, float* __restrict__ agg)
{
    __shared__ __bf16 xsT[8][64 * 32];      // per-wave transposed xs tile [j][edge]
    __shared__ __bf16 Bbuf[2][64 * BROW];   // double-buffered B rows for current j
    const int lane  = threadIdx.x & 31;
    const int wave  = threadIdx.x >> 5;
    const int l16   = lane & 15;
    const int off   = (lane < 16) ? 0 : 8;  // K-octet offset per ISA A/B layout
    const int ebase = (blockIdx.x * 8 + wave) * 32;

    // B stage: 64 rows (n=0..63) of 128 bf16 = 16KB; 256 threads x 64B each.
    const int srow = threadIdx.x >> 2;       // row n = 0..63
    const int scol = (threadIdx.x & 3) * 32; // 64B chunk within row
    auto stage = [&](int j, int parity) {
        const __bf16* gsrc = Bw + (size_t)(srow * 64 + j) * HID + scol;
        __bf16* ldst = &Bbuf[parity][srow * BROW + scol];
#if HAVE_ASYNC_LDS
#pragma unroll
        for (int q = 0; q < 4; ++q)
            __builtin_amdgcn_global_load_async_to_lds_b128(
                (gv4i_p)(gsrc + q * 8), (lv4i_p)(ldst + q * 8), 0, 0);
#else
#pragma unroll
        for (int q = 0; q < 4; ++q)
            *(v8bf*)(ldst + q * 8) = *(const v8bf*)(gsrc + q * 8);
#endif
    };

    // stage this wave's 32 xs rows, transposed (conflict-free: all lanes same j)
    {
        const __bf16* row = xsbf + (size_t)(ebase + lane) * NODE_DIM;
        __bf16 tmp[64];
#pragma unroll
        for (int q = 0; q < 8; ++q) {
            v8bf v = *(const v8bf*)(row + q * 8);
#pragma unroll
            for (int b = 0; b < 8; ++b) tmp[q * 8 + b] = v[b];
        }
#pragma unroll
        for (int j = 0; j < 64; ++j) xsT[wave][j * 32 + lane] = tmp[j];
    }

    stage(0, 0);   // prologue B stage for j=0

    // register-resident h fragments: lane holds its half of each K=32 chunk
    v16bf hreg[2][4];
#pragma unroll
    for (int m = 0; m < 2; ++m) {
        const __bf16* hrow = hbf + (size_t)(ebase + m * 16 + l16) * HID + off;
#pragma unroll
        for (int kc = 0; kc < 4; ++kc)
            hreg[m][kc] = combine16(hrow + kc * 32, hrow + kc * 32 + 16);
    }

    v8f acc[2][4];
#pragma unroll
    for (int m = 0; m < 2; ++m)
#pragma unroll
        for (int t = 0; t < 4; ++t) acc[m][t] = (v8f)(0.0f);

    // main reduction: j = 0..63 (x_src dim), kc = 0..3 (h dim, chunks of 32)
    for (int j = 0; j < 64; ++j) {
        wait_async0();        // my stage(j) writes have landed in LDS
        __syncthreads();      // all waves' stage(j) visible; buf[(j+1)&1] free
        if (j + 1 < 64) stage(j + 1, (j + 1) & 1);   // overlap with compute

        const __bf16* bb = &Bbuf[j & 1][0];
        float s0 = (float)xsT[wave][j * 32 + l16];
        float s1 = (float)xsT[wave][j * 32 + 16 + l16];
#pragma unroll
        for (int kc = 0; kc < 4; ++kc) {
            v16bf a0 = scale_bf(hreg[0][kc], s0);    // A = xs[e,j] * h[e, kchunk]
            v16bf a1 = scale_bf(hreg[1][kc], s1);
#pragma unroll
            for (int t = 0; t < 4; ++t) {
                const __bf16* bp = bb + (t * 16 + l16) * BROW + kc * 32 + off;
                v16bf bfr = combine16(bp, bp + 16);  // ds_load_b128 x2
                acc[0][t] = __builtin_amdgcn_wmma_f32_16x16x32_bf16(
                    false, a0, false, bfr, (short)0, acc[0][t], false, false);
                acc[1][t] = __builtin_amdgcn_wmma_f32_16x16x32_bf16(
                    false, a1, false, bfr, (short)0, acc[1][t], false, false);
            }
        }
    }

    // bias term: 2 extra K=32 chunks with A = xs directly, B = b2 (tiny, global)
#pragma unroll
    for (int jc = 0; jc < 2; ++jc) {
        v16bf a0, a1;
        {
            const __bf16* xr0 = xsbf + (size_t)(ebase + l16) * NODE_DIM + jc * 32 + off;
            const __bf16* xr1 = xsbf + (size_t)(ebase + 16 + l16) * NODE_DIM + jc * 32 + off;
            a0 = combine16(xr0, xr0 + 16);
            a1 = combine16(xr1, xr1 + 16);
        }
#pragma unroll
        for (int t = 0; t < 4; ++t) {
            const __bf16* bp = Bb2 + (size_t)(t * 16 + l16) * NODE_DIM + jc * 32 + off;
            v16bf bfr = combine16(bp, bp + 16);
            acc[0][t] = __builtin_amdgcn_wmma_f32_16x16x32_bf16(
                false, a0, false, bfr, (short)0, acc[0][t], false, false);
            acc[1][t] = __builtin_amdgcn_wmma_f32_16x16x32_bf16(
                false, a1, false, bfr, (short)0, acc[1][t], false, false);
        }
    }

    // scatter-add into agg (C/D layout: VGPR r -> M = r + off, N = lane%16)
#pragma unroll
    for (int m = 0; m < 2; ++m)
#pragma unroll
        for (int r = 0; r < 8; ++r) {
            int e = ebase + m * 16 + r + off;
            if (e < N_EDGES) {
                int d = dstIdx[e];
                float* orow = agg + (size_t)d * NODE_DIM + l16;
#pragma unroll
                for (int t = 0; t < 4; ++t)
                    __hip_atomic_fetch_add(orow + t * 16, acc[m][t][r],
                                           __ATOMIC_RELAXED, __HIP_MEMORY_SCOPE_AGENT);
            }
        }
}

// ---------------- k4: GRU cell, in-place over d_out (agg -> final) -------
__global__ __launch_bounds__(64) void gru_kernel(
    const float* __restrict__ x,
    const float* __restrict__ wih, const float* __restrict__ whh,
    const float* __restrict__ bih, const float* __restrict__ bhh,
    float* __restrict__ out)
{
    int v = blockIdx.x * blockDim.x + threadIdx.x;
    if (v >= N_NODES) return;
    float a[64], h[64];
#pragma unroll
    for (int k = 0; k < 64; ++k) a[k] = out[(size_t)v * 64 + k];
#pragma unroll
    for (int k = 0; k < 64; ++k) h[k] = x[(size_t)v * 64 + k];

    for (int i = 0; i < 64; ++i) {
        float ir = bih[i], iz = bih[64 + i], inn = bih[128 + i];
        float hr = bhh[i], hz = bhh[64 + i], hnn = bhh[128 + i];
        const float* wr = wih + (size_t)i * 64;
        const float* wz = wih + (size_t)(64 + i) * 64;
        const float* wn = wih + (size_t)(128 + i) * 64;
        const float* vr = whh + (size_t)i * 64;
        const float* vz = whh + (size_t)(64 + i) * 64;
        const float* vn = whh + (size_t)(128 + i) * 64;
#pragma unroll
        for (int k = 0; k < 64; ++k) {
            ir += wr[k] * a[k];  iz += wz[k] * a[k];  inn += wn[k] * a[k];
            hr += vr[k] * h[k];  hz += vz[k] * h[k];  hnn += vn[k] * h[k];
        }
        float r = 1.0f / (1.0f + __expf(-(ir + hr)));
        float z = 1.0f / (1.0f + __expf(-(iz + hz)));
        float n = tanhf(inn + r * hnn);
        out[(size_t)v * 64 + i] = (1.0f - z) * n + z * h[i];
    }
}

// ---------------- launcher ----------------
extern "C" void kernel_launch(void* const* d_in, const int* in_sizes, int n_in,
                              void* d_out, int out_size, void* d_ws, size_t ws_size,
                              hipStream_t stream) {
    const float* x   = (const float*)d_in[0];
    const int*   ei  = (const int*)d_in[1];      // (2, E): row0=src, row1=dst
    const float* ea  = (const float*)d_in[2];
    const float* W1  = (const float*)d_in[3];
    const float* b1  = (const float*)d_in[4];
    const float* W2  = (const float*)d_in[5];
    const float* b2  = (const float*)d_in[6];
    const float* wih = (const float*)d_in[7];
    const float* whh = (const float*)d_in[8];
    const float* bih = (const float*)d_in[9];
    const float* bhh = (const float*)d_in[10];
    float* out = (float*)d_out;

    char* ws = (char*)d_ws;
    __bf16* Bw   = (__bf16*)(ws + 0);                       // 4096*128 bf16 = 1 MB
    __bf16* Bb2  = (__bf16*)(ws + 1048576);                 // 4096 bf16
    __bf16* hbf  = (__bf16*)(ws + 1056768);                 // E_PAD*128 bf16
    __bf16* xsbf = (__bf16*)(ws + 1056768 + (size_t)E_PAD * HID * 2);

    const int* src = ei;
    const int* dst = ei + N_EDGES;

    zero_kernel<<<(N_NODES * NODE_DIM + 255) / 256, 256, 0, stream>>>(out, N_NODES * NODE_DIM);
    repack_kernel<<<(NODE_DIM * NODE_DIM * HID) / 256, 256, 0, stream>>>(W2, b2, Bw, Bb2);
    hcalc_kernel<<<(E_PAD * HID) / 256, 256, 0, stream>>>(ea, W1, b1, hbf);
    xgather_kernel<<<(E_PAD * NODE_DIM) / 256, 256, 0, stream>>>(x, src, xsbf);
    msg_kernel<<<E_PAD / 256, 256, 0, stream>>>(hbf, xsbf, Bw, Bb2, dst, out);
    gru_kernel<<<(N_NODES + 63) / 64, 64, 0, stream>>>(x, wih, whh, bih, bhh, out);
}
